// DecoderBlock_38534446580372
// MI455X (gfx1250) — matrix operand order repaired and checked by reading, not compiled, not run on hardware
//
#include <hip/hip_runtime.h>

// ---------------------------------------------------------------------------
// Types for CDNA5 WMMA (wave32): bf16 16x16x32, f32 accumulate.
// ---------------------------------------------------------------------------
typedef __attribute__((ext_vector_type(8)))  __bf16 v8bf;
typedef __attribute__((ext_vector_type(16))) __bf16 v16bf;
typedef __attribute__((ext_vector_type(8)))  float  v8f;
typedef __attribute__((ext_vector_type(4)))  unsigned int v4u;
typedef __attribute__((ext_vector_type(4)))  int    v4i;
typedef __attribute__((ext_vector_type(8)))  int    v8i;

#if __has_builtin(__builtin_amdgcn_tensor_load_to_lds)
#define USE_TDM 1
#else
#define USE_TDM 0
#endif

__device__ __forceinline__ v8f wmma_bf16(v16bf a, v16bf b, v8f c) {
  // (neg_a, A, neg_b, B, c_mod, C, reuse_a, reuse_b)
  return __builtin_amdgcn_wmma_f32_16x16x32_bf16(false, a, false, b,
                                                 (short)0, c, false, false);
}

// Load 16 contiguous bf16 (B-fragment style: lane<16 -> col n=lane%16,K=0..15;
// lane>=16 -> K=16..31; caller supplies the per-lane base pointer).
__device__ __forceinline__ v16bf ld16(const __bf16* p) {
  v8bf a = *(const v8bf*)p;
  v8bf b = *(const v8bf*)(p + 8);
  v16bf r;
#pragma unroll
  for (int i = 0; i < 8; ++i) { r[i] = a[i]; r[i + 8] = b[i]; }
  return r;
}

// A-fragment: row = lane%16; lane<16 holds K {0..15}, lane>=16 holds
// K {16..31}. rowp points at the 32-wide K row; kb = (lane>>4)*8.
__device__ __forceinline__ v16bf ld_a(const __bf16* rowp, int kb) {
  v8bf a = *(const v8bf*)(rowp + 2 * kb);
  v8bf b = *(const v8bf*)(rowp + 2 * kb + 8);
  v16bf r;
#pragma unroll
  for (int i = 0; i < 8; ++i) { r[i] = a[i]; r[i + 8] = b[i]; }
  return r;
}

// ---------------------------------------------------------------------------
// Generic bf16 GEMM: C[M,N] = A[M,K] @ B[K,N] (+bias, optional ReLU).
// WG = 256 threads = 8 waves (2 in N, 4 in M). WG tile 128x128, K-step 32.
// A tile staged into LDS via the Tensor Data Mover (2D D# descriptor,
// tile 32(K) x 128(M) of 2-byte elements), B staged transposed [128n][32k]
// by VALU so B-fragments are contiguous 32B ds reads.
// ---------------------------------------------------------------------------
template <bool RELU>
__global__ __launch_bounds__(256) void gemm_bf16_kernel(
    const __bf16* __restrict__ A, const __bf16* __restrict__ B,
    const float* __restrict__ bias,
    __bf16* __restrict__ Cb, float* __restrict__ Cf,
    int M, int N, int K) {
  __shared__ alignas(32) __bf16 As[128 * 32];
  __shared__ alignas(32) __bf16 Bs[128 * 32];
  const int t = threadIdx.x;
  const int wid = t >> 5, lane = t & 31;
  const int wm = wid & 3, wn = wid >> 2;
  const int m_blk = blockIdx.y * 128, n_blk = blockIdx.x * 128;
  const int rlo = lane & 15;
  const int kb8 = (lane >> 4) * 8;
  const int kk16 = (lane >> 4) * 16;

  v8f acc[2][4];
#pragma unroll
  for (int i = 0; i < 2; ++i)
#pragma unroll
    for (int j = 0; j < 4; ++j)
#pragma unroll
      for (int r = 0; r < 8; ++r) acc[i][j][r] = 0.0f;

  for (int k0 = 0; k0 < K; k0 += 32) {
#if USE_TDM
    if (wid == 0) {
      // TDM D# for the A tile: rows m_blk..m_blk+127, cols k0..k0+31 (bf16).
      unsigned long long ga =
          (unsigned long long)(size_t)(A + (size_t)m_blk * K + k0);
      unsigned lds0 = (unsigned)(size_t)(&As[0]);
      v4u g0;
      g0[0] = 1u;                       // count=1, user descriptor
      g0[1] = lds0;                     // lds_addr (bytes)
      g0[2] = (unsigned)ga;             // global_addr[31:0]
      g0[3] = (unsigned)((ga >> 32) & 0x01ffffffu) | (2u << 30);  // type=2
      v8i g1;
      g1[0] = (int)(1u << 16);          // data_size=1 -> 2 bytes/elem
      g1[1] = (int)(((unsigned)K & 0xffffu) << 16);        // tensor_dim0 lo
      g1[2] = (int)(((unsigned)K >> 16) |
                    (((unsigned)M & 0xffffu) << 16));      // dim0 hi | dim1 lo
      g1[3] = (int)(((unsigned)M >> 16) | (32u << 16));    // dim1 hi | tile_dim0
      g1[4] = 128;                      // tile_dim1=128, tile_dim2=0
      g1[5] = K;                        // tensor_dim0_stride lo (elems)
      g1[6] = 0;                        // stride hi | dim1_stride lo
      g1[7] = 0;
      v4i gz = {0, 0, 0, 0};
#if defined(__clang_major__) && (__clang_major__ >= 23)
      v8i g4 = {0, 0, 0, 0, 0, 0, 0, 0};
      __builtin_amdgcn_tensor_load_to_lds(g0, g1, gz, gz, g4, 0);
#else
      __builtin_amdgcn_tensor_load_to_lds(g0, g1, gz, gz, 0);
#endif
      __builtin_amdgcn_s_wait_tensorcnt(0);
    }
#else
    {  // fallback staging: each thread moves 16 contiguous bf16
      int row = t >> 1, off = (t & 1) * 16;
      const __bf16* g = A + (size_t)(m_blk + row) * K + k0 + off;
      *(v8bf*)&As[row * 32 + off]     = *(const v8bf*)g;
      *(v8bf*)&As[row * 32 + off + 8] = *(const v8bf*)(g + 8);
    }
#endif
    {  // stage B tile 32x128 transposed into LDS [n][k]
      int krow = t >> 3, noff = (t & 7) * 16;
      const __bf16* g = B + (size_t)(k0 + krow) * N + n_blk + noff;
      v8bf x0 = *(const v8bf*)g;
      v8bf x1 = *(const v8bf*)(g + 8);
#pragma unroll
      for (int i = 0; i < 8; ++i) {
        Bs[(noff + i) * 32 + krow]     = x0[i];
        Bs[(noff + 8 + i) * 32 + krow] = x1[i];
      }
    }
    if (k0 + 32 < K) {  // prefetch next B K-tile (global_prefetch_b8)
      __builtin_prefetch(B + (size_t)(k0 + 32 + (t >> 3)) * N + n_blk + (t & 7) * 16, 0, 1);
    }
    __syncthreads();

    v16bf af[2], bfr[4];
#pragma unroll
    for (int i = 0; i < 2; ++i)
      af[i] = ld_a(&As[(wm * 32 + i * 16 + rlo) * 32], kb8);
#pragma unroll
    for (int j = 0; j < 4; ++j)
      bfr[j] = ld16(&Bs[(wn * 64 + j * 16 + rlo) * 32 + kk16]);
#pragma unroll
    for (int i = 0; i < 2; ++i)
#pragma unroll
      for (int j = 0; j < 4; ++j)
        acc[i][j] = wmma_bf16(af[i], bfr[j], acc[i][j]);
    __syncthreads();
  }

  // Epilogue: C/D layout -> lane holds col (n_frag+lane%16), rows r+8*(lane/16)
  const int colb = n_blk + wn * 64;
  const int rowb = m_blk + wm * 32;
#pragma unroll
  for (int i = 0; i < 2; ++i) {
#pragma unroll
    for (int j = 0; j < 4; ++j) {
      int col = colb + j * 16 + rlo;
      float bv = bias ? bias[col] : 0.0f;
      int r0 = rowb + i * 16 + ((lane >> 4) * 8);
#pragma unroll
      for (int r = 0; r < 8; ++r) {
        float v = acc[i][j][r] + bv;
        if (RELU) v = fmaxf(v, 0.0f);
        size_t idx = (size_t)(r0 + r) * N + col;
        if (Cb) Cb[idx] = (__bf16)v;
        if (Cf) Cf[idx] = v;
      }
    }
  }
}

// ---------------------------------------------------------------------------
// Flash attention (causal), one wave per 16 q-rows. qkv: [B*T, 1536] bf16
// (cols 0..511=Q, 512..1023=K, 1024..1535=V, head-major). vT: [B*H][64][512].
// ---------------------------------------------------------------------------
__global__ __launch_bounds__(128) void attn_kernel(
    const __bf16* __restrict__ qkv, const __bf16* __restrict__ vT,
    __bf16* __restrict__ attn_o) {
  const int T = 512, D3 = 1536;
  const int wg = blockIdx.x;
  const int tile = wg & 7;      // 8 tiles of 64 rows each
  const int bh = wg >> 3;       // 0..255
  const int b = bh >> 3, h = bh & 7;
  const int wave = threadIdx.x >> 5, lane = threadIdx.x & 31;
  const int m0 = tile * 64 + wave * 16;

  const __bf16* qb  = qkv + (size_t)b * T * D3 + h * 64;
  const __bf16* kbp = qkv + (size_t)b * T * D3 + 512 + h * 64;
  const __bf16* vtb = vT + (size_t)bh * 64 * T;
  __shared__ alignas(32) __bf16 pshm[4][16 * 32];
  __bf16* pw = pshm[wave];

  const int rlo = lane & 15;
  const int kb8 = (lane >> 4) * 8;
  const int kk16 = (lane >> 4) * 16;
  const int rowb = m0 + ((lane >> 4) * 8);

  v16bf aq0 = ld_a(qb + (size_t)(m0 + rlo) * D3, kb8);        // K 0..31
  v16bf aq1 = ld_a(qb + (size_t)(m0 + rlo) * D3 + 32, kb8);   // K 32..63

  float mrun[8], lrun[8];
  v8f o[4];
#pragma unroll
  for (int r = 0; r < 8; ++r) { mrun[r] = -3.0e38f; lrun[r] = 0.0f; }
#pragma unroll
  for (int f = 0; f < 4; ++f)
#pragma unroll
    for (int r = 0; r < 8; ++r) o[f][r] = 0.0f;

  const float scale = 0.125f;  // 1/sqrt(64)

  for (int s0 = 0; s0 < m0 + 16; s0 += 32) {
    v8f sc0, sc1;
#pragma unroll
    for (int r = 0; r < 8; ++r) { sc0[r] = 0.0f; sc1[r] = 0.0f; }
    {  // scores: q[16x64] @ k^T[64x32] as 2 n-subtiles x 2 k-steps
      v16bf bk0 = ld16(kbp + (size_t)(s0 + rlo) * D3 + kk16);
      v16bf bk1 = ld16(kbp + (size_t)(s0 + 16 + rlo) * D3 + kk16);
      sc0 = wmma_bf16(aq0, bk0, sc0);
      sc1 = wmma_bf16(aq0, bk1, sc1);
      v16bf bk2 = ld16(kbp + (size_t)(s0 + rlo) * D3 + 32 + kk16);
      v16bf bk3 = ld16(kbp + (size_t)(s0 + 16 + rlo) * D3 + 32 + kk16);
      sc0 = wmma_bf16(aq1, bk2, sc0);
      sc1 = wmma_bf16(aq1, bk3, sc1);
    }
    const int col0 = s0 + rlo;
    float p0[8], p1[8], cm[8];
#pragma unroll
    for (int r = 0; r < 8; ++r) {
      int ra = rowb + r;
      float x0 = sc0[r] * scale; if (col0 > ra)      x0 = -3.0e38f;
      float x1 = sc1[r] * scale; if (col0 + 16 > ra) x1 = -3.0e38f;
      p0[r] = x0; p1[r] = x1; cm[r] = fmaxf(x0, x1);
    }
#pragma unroll
    for (int off = 1; off < 16; off <<= 1)
#pragma unroll
      for (int r = 0; r < 8; ++r)
        cm[r] = fmaxf(cm[r], __shfl_xor(cm[r], off));
    float rs[8];
#pragma unroll
    for (int r = 0; r < 8; ++r) {
      float mnew = fmaxf(mrun[r], cm[r]);
      float al = __expf(mrun[r] - mnew);
      p0[r] = __expf(p0[r] - mnew);
      p1[r] = __expf(p1[r] - mnew);
      rs[r] = p0[r] + p1[r];
      lrun[r] *= al;
      mrun[r] = mnew;
#pragma unroll
      for (int f = 0; f < 4; ++f) o[f][r] *= al;
    }
#pragma unroll
    for (int off = 1; off < 16; off <<= 1)
#pragma unroll
      for (int r = 0; r < 8; ++r)
        rs[r] += __shfl_xor(rs[r], off);
#pragma unroll
    for (int r = 0; r < 8; ++r) lrun[r] += rs[r];

    // C-layout P -> LDS -> A-layout fragment (per-wave private region)
#pragma unroll
    for (int r = 0; r < 8; ++r) {
      int rr = r + ((lane >> 4) * 8);
      pw[rr * 32 + rlo]      = (__bf16)p0[r];
      pw[rr * 32 + 16 + rlo] = (__bf16)p1[r];
    }
    asm volatile("s_wait_dscnt 0x0" ::: "memory");
    v16bf pa = ld_a(&pw[rlo * 32], kb8);
#pragma unroll
    for (int f = 0; f < 4; ++f) {
      v16bf bv = ld16(vtb + (size_t)(f * 16 + rlo) * T + s0 + kk16);
      o[f] = wmma_bf16(pa, bv, o[f]);
    }
  }

  __bf16* ob = attn_o + (size_t)b * T * 512 + h * 64;
#pragma unroll
  for (int r = 0; r < 8; ++r) {
    float inv = 1.0f / lrun[r];
    int ra = rowb + r;
#pragma unroll
    for (int f = 0; f < 4; ++f)
      ob[(size_t)ra * 512 + f * 16 + rlo] = (__bf16)(o[f][r] * inv);
  }
}

// ---------------------------------------------------------------------------
// out = base + LayerNorm(y)*g + beta ; optional bf16 mirror for next GEMM.
// One block (256 thr) per 512-elem row: wave shuffle + LDS reduce.
// ---------------------------------------------------------------------------
__global__ __launch_bounds__(256) void ln_add_kernel(
    const float* __restrict__ base, const float* __restrict__ y,
    const float* __restrict__ g, const float* __restrict__ beta,
    float* __restrict__ outf, __bf16* __restrict__ outb) {
  const int row = blockIdx.x, t = threadIdx.x;
  const size_t roff = (size_t)row * 512;
  float v0 = y[roff + t], v1 = y[roff + t + 256];
  __shared__ float wred[8];
  float s = v0 + v1;
#pragma unroll
  for (int off = 16; off > 0; off >>= 1) s += __shfl_xor(s, off);
  if ((t & 31) == 0) wred[t >> 5] = s;
  __syncthreads();
  float mu = 0.0f;
#pragma unroll
  for (int i = 0; i < 8; ++i) mu += wred[i];
  mu *= (1.0f / 512.0f);
  __syncthreads();
  float d0 = v0 - mu, d1 = v1 - mu;
  float q = d0 * d0 + d1 * d1;
#pragma unroll
  for (int off = 16; off > 0; off >>= 1) q += __shfl_xor(q, off);
  if ((t & 31) == 0) wred[t >> 5] = q;
  __syncthreads();
  float var = 0.0f;
#pragma unroll
  for (int i = 0; i < 8; ++i) var += wred[i];
  var *= (1.0f / 512.0f);
  float rstd = rsqrtf(var + 1e-5f);
  float o0 = base[roff + t]       + d0 * rstd * g[t]       + beta[t];
  float o1 = base[roff + t + 256] + d1 * rstd * g[t + 256] + beta[t + 256];
  outf[roff + t] = o0;
  outf[roff + t + 256] = o1;
  if (outb) {
    outb[roff + t] = (__bf16)o0;
    outb[roff + t + 256] = (__bf16)o1;
  }
}

// ---------------------------------------------------------------------------
// Helpers: f32->bf16 convert; fuse per-head Wq/Wk/Wv into [512,1536] bf16
// (+ fused bias); V transpose [B*T,512 v-cols] -> [B*H][64][512].
// ---------------------------------------------------------------------------
__global__ void cvt_bf16_kernel(const float* __restrict__ in,
                                __bf16* __restrict__ out, int n) {
  int i = blockIdx.x * 256 + threadIdx.x;
  if (i < n) out[i] = (__bf16)in[i];
}

__global__ void build_wqkv_kernel(
    const float* __restrict__ Wq, const float* __restrict__ Wk,
    const float* __restrict__ Wv, const float* __restrict__ bq,
    const float* __restrict__ bk, const float* __restrict__ bv,
    __bf16* __restrict__ W, float* __restrict__ bias) {
  int idx = blockIdx.x * 256 + threadIdx.x;  // 512*1536
  if (idx >= 512 * 1536) return;
  int d = idx / 1536, c = idx % 1536;
  int sec = c >> 9, ce = c & 511, h = ce >> 6, e = ce & 63;
  const float* Ws = (sec == 0) ? Wq : ((sec == 1) ? Wk : Wv);
  W[idx] = (__bf16)Ws[((size_t)h * 512 + d) * 64 + e];
  if (d == 0) {
    const float* bs = (sec == 0) ? bq : ((sec == 1) ? bk : bv);
    bias[c] = bs[h * 64 + e];
  }
}

__global__ void transpose_v_kernel(const __bf16* __restrict__ qkv,
                                   __bf16* __restrict__ vT) {
  int idx = blockIdx.x * 256 + threadIdx.x;  // 256*64*512
  if (idx >= 256 * 64 * 512) return;
  int tk = idx & 511;
  int r = idx >> 9;
  int e = r & 63; r >>= 6;
  int h = r & 7;
  int b = r >> 3;
  vT[idx] = qkv[((size_t)(b * 512 + tk)) * 1536 + 1024 + h * 64 + e];
}

// ---------------------------------------------------------------------------
extern "C" void kernel_launch(void* const* d_in, const int* in_sizes, int n_in,
                              void* d_out, int out_size, void* d_ws, size_t ws_size,
                              hipStream_t stream) {
  const float* x    = (const float*)d_in[0];
  const float* Wq   = (const float*)d_in[1];
  const float* bq   = (const float*)d_in[2];
  const float* Wk   = (const float*)d_in[3];
  const float* bk   = (const float*)d_in[4];
  const float* Wv   = (const float*)d_in[5];
  const float* bv   = (const float*)d_in[6];
  const float* Wo   = (const float*)d_in[7];
  const float* bo   = (const float*)d_in[8];
  const float* ln1g = (const float*)d_in[9];
  const float* ln1b = (const float*)d_in[10];
  const float* W1   = (const float*)d_in[11];
  const float* b1   = (const float*)d_in[12];
  const float* W2   = (const float*)d_in[13];
  const float* b2   = (const float*)d_in[14];
  const float* ln2g = (const float*)d_in[15];
  const float* ln2b = (const float*)d_in[16];
  float* out = (float*)d_out;

  const int Mtok = 16384, D = 512, D3 = 1536, DFF = 2048;
  char* wsb = (char*)d_ws;
  size_t off = 0;
  auto alloc = [&](size_t bytes) -> void* {
    void* p = wsb + off;
    off += (bytes + 255) & ~(size_t)255;
    return p;
  };
  __bf16* xb    = (__bf16*)alloc((size_t)Mtok * D * 2);
  __bf16* Wqkvb = (__bf16*)alloc((size_t)D * D3 * 2);
  float*  bqkv  = (float*) alloc((size_t)D3 * 4);
  __bf16* qkvb  = (__bf16*)alloc((size_t)Mtok * D3 * 2);
  __bf16* vTb   = (__bf16*)alloc((size_t)256 * 64 * 512 * 2);
  __bf16* aob   = (__bf16*)alloc((size_t)Mtok * D * 2);
  __bf16* Wob   = (__bf16*)alloc((size_t)D * D * 2);
  float*  aprj  = (float*) alloc((size_t)Mtok * D * 4);
  float*  out1f = (float*) alloc((size_t)Mtok * D * 4);
  __bf16* out1b = (__bf16*)alloc((size_t)Mtok * D * 2);
  __bf16* W1b   = (__bf16*)alloc((size_t)D * DFF * 2);
  __bf16* W2b   = (__bf16*)alloc((size_t)DFF * D * 2);
  __bf16* hb    = (__bf16*)alloc((size_t)Mtok * DFF * 2);
  float*  ffo   = (float*) alloc((size_t)Mtok * D * 4);

  cvt_bf16_kernel<<<(Mtok * D) / 256, 256, 0, stream>>>(x, xb, Mtok * D);
  build_wqkv_kernel<<<(D * D3) / 256, 256, 0, stream>>>(Wq, Wk, Wv, bq, bk, bv,
                                                        Wqkvb, bqkv);
  cvt_bf16_kernel<<<(D * D) / 256, 256, 0, stream>>>(Wo, Wob, D * D);
  cvt_bf16_kernel<<<(D * DFF) / 256, 256, 0, stream>>>(W1, W1b, D * DFF);
  cvt_bf16_kernel<<<(DFF * D) / 256, 256, 0, stream>>>(W2, W2b, DFF * D);

  // fused QKV projection: [16384,512] @ [512,1536] (+bias) -> bf16
  gemm_bf16_kernel<false><<<dim3(D3 / 128, Mtok / 128), 256, 0, stream>>>(
      xb, Wqkvb, bqkv, qkvb, nullptr, Mtok, D3, D);
  transpose_v_kernel<<<(256 * 64 * 512) / 256, 256, 0, stream>>>(qkvb, vTb);
  attn_kernel<<<2048, 128, 0, stream>>>(qkvb, vTb, aob);
  // output projection -> f32 (pre-LN value)
  gemm_bf16_kernel<false><<<dim3(D / 128, Mtok / 128), 256, 0, stream>>>(
      aob, Wob, bo, nullptr, aprj, Mtok, D, D);
  ln_add_kernel<<<Mtok, 256, 0, stream>>>(x, aprj, ln1g, ln1b, out1f, out1b);
  // FFN
  gemm_bf16_kernel<true><<<dim3(DFF / 128, Mtok / 128), 256, 0, stream>>>(
      out1b, W1b, b1, hb, nullptr, Mtok, DFF, D);
  gemm_bf16_kernel<false><<<dim3(D / 128, Mtok / 128), 256, 0, stream>>>(
      hb, W2b, b2, nullptr, ffo, Mtok, D, DFF);
  ln_add_kernel<<<Mtok, 256, 0, stream>>>(out1f, ffo, ln2g, ln2b, out, nullptr);

  (void)in_sizes; (void)n_in; (void)out_size; (void)ws_size;
}
